// Decoder_5909874999389
// MI455X (gfx1250) — compile-verified, matrix operand order checked by reading, not run on hardware
//
#include <hip/hip_runtime.h>

// ---------------------------------------------------------------------------
// Problem constants (from reference setup)
// ---------------------------------------------------------------------------
constexpr int kB   = 64;     // batch
constexpr int kP   = 196;    // pixels
constexpr int kE   = 2048;   // encoder dim
constexpr int kD   = 512;    // decoder dim
constexpr int kA   = 512;    // attention dim
constexpr int kM   = 512;    // embedding dim
constexpr int kV   = 10000;  // vocab
constexpr int kT   = 30;     // max(decode_lengths - 1) given dl[0]=31
constexpr int kCap = 31;
constexpr int kXH  = kM + kE + kD;   // 3072 : [emb | gated awe | h]
constexpr int kG   = 4 * kD;         // 2048 gates
constexpr int kKS  = 4;              // split-K factor for the gates GEMM

// d_out layout: predictions (B,T,V) | dl (B) | alphas (B,T,P) | inds (B)
constexpr size_t OFF_DL    = (size_t)kB * kT * kV;
constexpr size_t OFF_ALPHA = OFF_DL + kB;
constexpr size_t OFF_INDS  = OFF_ALPHA + (size_t)kB * kT * kP;

// ---------------------------------------------------------------------------
// Helpers
// ---------------------------------------------------------------------------
typedef __attribute__((ext_vector_type(16))) __bf16 v16bf;
typedef __attribute__((ext_vector_type(8)))  float  v8f;

struct __align__(16) U128 { unsigned int x, y, z, w; };
union BFrag { v16bf v; U128 u[2]; };

__device__ __forceinline__ unsigned short f2bf(float f) {
    unsigned int u = __builtin_bit_cast(unsigned int, f);
    u += 0x7FFFu + ((u >> 16) & 1u);          // round-to-nearest-even
    return (unsigned short)(u >> 16);
}
__device__ __forceinline__ float bf2f(unsigned short h) {
    unsigned int u = ((unsigned int)h) << 16;
    return __builtin_bit_cast(float, u);
}
__device__ __forceinline__ float sigmoidf(float x) {
    return 1.0f / (1.0f + __expf(-x));
}
__device__ __forceinline__ v8f wmma_bf16(const BFrag& a, const BFrag& b, v8f c) {
    return __builtin_amdgcn_wmma_f32_16x16x32_bf16(false, a.v, false, b.v,
                                                   (short)0, c, false, false);
}

// ---------------------------------------------------------------------------
// Generic bf16 WMMA GEMM (big-M path, used for the one-time enc_att GEMM):
//   C[M,N] = A[M,K] * Bt[N,K]^T + bias ; wave computes 16x64, A frag reused.
//   Subtile guards scalarized via readfirstlane -> s_cbranch, no EXEC churn.
// ---------------------------------------------------------------------------
template <bool OUT_BF16>
__global__ __launch_bounds__(256) void k_gemm_bf16(
    const unsigned short* __restrict__ A,  int lda,
    const unsigned short* __restrict__ Bt, int ldb,
    const float* __restrict__ bias,
    void* __restrict__ Cout, int ldc,
    int N, int K)
{
    const int tid  = threadIdx.x;
    const int wave = tid >> 5;
    const int lane = tid & 31;
    const int l16  = lane & 15;
    const int kh   = lane >> 4;
    const int m0   = blockIdx.x << 4;
    const int n_base = (blockIdx.y * 8 + wave) * 64;

    if (__builtin_amdgcn_readfirstlane((int)((n_base + 16) <= N)) == 0) return;
    const int v1 = __builtin_amdgcn_readfirstlane((int)((n_base + 32) <= N));
    const int v2 = __builtin_amdgcn_readfirstlane((int)((n_base + 48) <= N));
    const int v3 = __builtin_amdgcn_readfirstlane((int)((n_base + 64) <= N));

    const unsigned short* arow = A + (size_t)(m0 + l16) * lda;
    const unsigned short* b0 = Bt + (size_t)(n_base +  0 + l16) * ldb;
    const unsigned short* b1 = Bt + (size_t)(n_base + 16 + l16) * ldb;
    const unsigned short* b2 = Bt + (size_t)(n_base + 32 + l16) * ldb;
    const unsigned short* b3 = Bt + (size_t)(n_base + 48 + l16) * ldb;

    v8f acc0 = 0.0f, acc1 = 0.0f, acc2 = 0.0f, acc3 = 0.0f;

    for (int k = 0; k < K; k += 32) {
        const int kb = k + kh * 8;
        if (k + 32 < K) __builtin_prefetch((const void*)(arow + kb + 32), 0, 3);
        BFrag a;
        a.u[0] = *(const U128*)(arow + kb);
        a.u[1] = *(const U128*)(arow + kb + 16);
        BFrag fb;
        fb.u[0] = *(const U128*)(b0 + kb);
        fb.u[1] = *(const U128*)(b0 + kb + 16);
        acc0 = wmma_bf16(a, fb, acc0);
        if (v1) {
            fb.u[0] = *(const U128*)(b1 + kb);
            fb.u[1] = *(const U128*)(b1 + kb + 16);
            acc1 = wmma_bf16(a, fb, acc1);
        }
        if (v2) {
            fb.u[0] = *(const U128*)(b2 + kb);
            fb.u[1] = *(const U128*)(b2 + kb + 16);
            acc2 = wmma_bf16(a, fb, acc2);
        }
        if (v3) {
            fb.u[0] = *(const U128*)(b3 + kb);
            fb.u[1] = *(const U128*)(b3 + kb + 16);
            acc3 = wmma_bf16(a, fb, acc3);
        }
    }

    const int rbase = m0 + kh * 8;
#define STORE_TILE(ACC, NSUB)                                              \
    {                                                                      \
        const int col = (NSUB) + l16;                                      \
        const float bv = bias ? bias[col] : 0.0f;                          \
        _Pragma("unroll")                                                  \
        for (int r = 0; r < 8; ++r) {                                      \
            const float val = (ACC)[r] + bv;                               \
            const size_t idx = (size_t)(rbase + r) * ldc + col;            \
            if (OUT_BF16) ((unsigned short*)Cout)[idx] = f2bf(val);        \
            else          ((float*)Cout)[idx] = val;                       \
        }                                                                  \
    }
    STORE_TILE(acc0, n_base)
    if (v1) STORE_TILE(acc1, n_base + 16)
    if (v2) STORE_TILE(acc2, n_base + 32)
    if (v3) STORE_TILE(acc3, n_base + 48)
#undef STORE_TILE
}

// ---------------------------------------------------------------------------
// Skinny-M GEMM for the recurrent steps (M == 64 == kB):
//   One block covers all 4 M-tiles; each wave owns one 16-column strip of N.
//   Per k-step: 1 B fragment (global, streamed, prefetched) feeds 4 WMMAs
//   against 4 A fragments read from a double-buffered LDS stage of the
//   64x32 A tile (80B row pitch -> conflict-free ds_load_b128).
//   Every weight byte is read exactly once per GEMM.
//   PARTIAL: split-K partials written to Cout + z*64*ldc (no bias); reduced
//   later by k_reduce (deterministic, no FP atomics).
//   No WMMA is conditional: OOB waves use clamped B rows, stores guarded.
// ---------------------------------------------------------------------------
template <bool PARTIAL>
__global__ __launch_bounds__(256) void k_gemm_skinny(
    const unsigned short* __restrict__ A,  int lda,   // kB x K row-major bf16
    const unsigned short* __restrict__ Bt, int ldb,   // N x K row-major bf16
    const float* __restrict__ bias,
    float* __restrict__ Cout, int ldc,
    int N, int K, int kchunk)
{
    constexpr int PITCH = 40;                      // elements; 80B row pitch
    __shared__ __align__(16) unsigned short sA[2][64 * PITCH];

    const int tid  = threadIdx.x;
    const int wave = tid >> 5;
    const int lane = tid & 31;
    const int l16  = lane & 15;
    const int kh   = lane >> 4;

    const int n0 = (blockIdx.x * 8 + wave) * 16;   // this wave's 16-col strip
    const int z  = blockIdx.y;                     // split-K index
    const int k0 = z * kchunk;
    const int kend = (k0 + kchunk < K) ? (k0 + kchunk) : K;
    const int nk = (kend - k0) >> 5;               // k-steps of 32

    const bool nvalid = (n0 + 16) <= N;            // wave-uniform at runtime
    const int  nclamp = nvalid ? n0 : 0;           // clamped: loads stay legal
    const unsigned short* brow = Bt + (size_t)(nclamp + l16) * ldb;

    // cooperative stage of one 64x32 A tile into LDS buffer `buf`
    const int srow = tid >> 2;                     // 64 rows
    const int schk = (tid & 3) * 8;                // 4 x 16B chunks per row
    auto stage = [&](int k, int buf) {
        U128 v = *(const U128*)(A + (size_t)srow * lda + k + schk);
        *(U128*)(&sA[buf][srow * PITCH + schk]) = v;
    };

    v8f acc0 = 0.0f, acc1 = 0.0f, acc2 = 0.0f, acc3 = 0.0f;
    stage(k0, 0);

    for (int i = 0; i < nk; ++i) {
        __syncthreads();                           // staged tile i visible
        if (i + 1 < nk) stage(k0 + (i + 1) * 32, (i + 1) & 1);

        const int kb = k0 + i * 32 + kh * 8;
        if (i + 1 < nk) __builtin_prefetch((const void*)(brow + kb + 32), 0, 3);
        BFrag fb;
        fb.u[0] = *(const U128*)(brow + kb);
        fb.u[1] = *(const U128*)(brow + kb + 16);

        const unsigned short* sb = &sA[i & 1][0];
        const int so = l16 * PITCH + kh * 8;
        BFrag a;
        a.u[0] = *(const U128*)(sb + so);
        a.u[1] = *(const U128*)(sb + so + 16);
        acc0 = wmma_bf16(a, fb, acc0);
        a.u[0] = *(const U128*)(sb + 16 * PITCH + so);
        a.u[1] = *(const U128*)(sb + 16 * PITCH + so + 16);
        acc1 = wmma_bf16(a, fb, acc1);
        a.u[0] = *(const U128*)(sb + 32 * PITCH + so);
        a.u[1] = *(const U128*)(sb + 32 * PITCH + so + 16);
        acc2 = wmma_bf16(a, fb, acc2);
        a.u[0] = *(const U128*)(sb + 48 * PITCH + so);
        a.u[1] = *(const U128*)(sb + 48 * PITCH + so + 16);
        acc3 = wmma_bf16(a, fb, acc3);
    }

    if (!nvalid) return;                           // stores only; no barriers left
    const int col = n0 + l16;
    const float bv = (!PARTIAL && bias) ? bias[col] : 0.0f;
    float* obase = Cout + (PARTIAL ? (size_t)z * 64 * ldc : (size_t)0);
#define STORE_MT(ACC, MT)                                                  \
    {                                                                      \
        const int rb = (MT) * 16 + kh * 8;                                 \
        _Pragma("unroll")                                                  \
        for (int r = 0; r < 8; ++r)                                        \
            obase[(size_t)(rb + r) * ldc + col] = (ACC)[r] + bv;           \
    }
    STORE_MT(acc0, 0)
    STORE_MT(acc1, 1)
    STORE_MT(acc2, 2)
    STORE_MT(acc3, 3)
#undef STORE_MT
}

// deterministic split-K reduction: out[i] = bias[i % ldc] + sum_z part[z][i]
__global__ void k_reduce(const float* __restrict__ part, const float* __restrict__ bias,
                         float* __restrict__ out, int n, int ldc, int zcount) {
    int i = blockIdx.x * 256 + threadIdx.x;
    if (i < n) {
        float s = bias ? bias[i % ldc] : 0.0f;
        for (int zz = 0; zz < zcount; ++zz) s += part[(size_t)zz * n + i];
        out[i] = s;
    }
}

// ---------------------------------------------------------------------------
// One-time conversion kernels
// ---------------------------------------------------------------------------
__global__ void k_f2bf(const float* __restrict__ in, unsigned short* __restrict__ out, int n) {
    int i = blockIdx.x * 256 + threadIdx.x;
    if (i < n) out[i] = f2bf(in[i]);
}

// in: R x C row-major fp32  ->  out: C x R row-major bf16
__global__ void k_transpose_f2bf(const float* __restrict__ in, unsigned short* __restrict__ out,
                                 int R, int C) {
    int i = blockIdx.x * 256 + threadIdx.x;
    if (i < R * C) {
        int r = i / C, c = i - r * C;
        out[(size_t)c * R + r] = f2bf(in[i]);
    }
}

// copy rows x cols fp32 into bf16 matrix with row stride ldo at column offset
__global__ void k_f2bf_sub(const float* __restrict__ in, unsigned short* __restrict__ out,
                           int rows, int cols, int ldo, int coff) {
    int i = blockIdx.x * 256 + threadIdx.x;
    if (i < rows * cols) {
        int r = i / cols, c = i - r * cols;
        out[(size_t)r * ldo + coff + c] = f2bf(in[i]);
    }
}

__global__ void k_vadd(const float* a, const float* b, float* o, int n) {
    int i = blockIdx.x * 256 + threadIdx.x;
    if (i < n) o[i] = a[i] + b[i];
}

// mean over pixels: grid (B, E/256)
__global__ void k_mean(const float* __restrict__ enc, float* __restrict__ mean) {
    int b = blockIdx.x;
    int e = blockIdx.y * 256 + threadIdx.x;
    const float* p = enc + (size_t)b * kP * kE + e;
    float s = 0.0f;
    for (int i = 0; i < kP; ++i) s += p[(size_t)i * kE];
    mean[b * kE + e] = s * (1.0f / kP);
}

// h0/c0 init: grid (B, D/256)
__global__ void k_init(const float* __restrict__ mean,
                       const float* __restrict__ hw, const float* __restrict__ hb,
                       const float* __restrict__ cw, const float* __restrict__ cb,
                       float* h, float* c, unsigned short* hbf) {
    int b = blockIdx.x;
    int d = blockIdx.y * 256 + threadIdx.x;
    const float* m = mean + b * kE;
    float sh = hb[d], sc = cb[d];
    for (int e = 0; e < kE; ++e) {
        float mv = m[e];
        sh += mv * hw[(size_t)e * kD + d];
        sc += mv * cw[(size_t)e * kD + d];
    }
    h[b * kD + d] = sh;
    c[b * kD + d] = sc;
    hbf[b * kD + d] = f2bf(sh);
}

__global__ void k_dlinds(const int* __restrict__ dlen, const int* __restrict__ inds,
                         float* __restrict__ out) {
    int b = threadIdx.x;
    if (b < kB) {
        out[OFF_DL + b]   = (float)(dlen[b] - 1);
        out[OFF_INDS + b] = (float)inds[b];
    }
}

// ---------------------------------------------------------------------------
// Per-step kernels
// ---------------------------------------------------------------------------
// attention scores + softmax; one block per batch row
__global__ __launch_bounds__(256) void k_att(
    const unsigned short* __restrict__ encatt,  // (B,P,A) bf16
    const float* __restrict__ dec,              // (B,A)
    const float* __restrict__ attw, const float* __restrict__ attb,
    const int* __restrict__ dlen, int t,
    float* __restrict__ alpha,                  // (B,P)
    float* __restrict__ out)                    // d_out (alphas region)
{
    __shared__ float sdec[kA];
    __shared__ float sw[kA];
    __shared__ float se[kP];
    const int b = blockIdx.x, tid = threadIdx.x;
    const int wave = tid >> 5, lane = tid & 31;

    for (int i = tid; i < kA; i += 256) { sdec[i] = dec[b * kA + i]; sw[i] = attw[i]; }
    __syncthreads();

    const unsigned short* ea = encatt + (size_t)b * kP * kA;
    for (int p = wave; p < kP; p += 8) {
        const unsigned short* row = ea + (size_t)p * kA;
        float s = 0.0f;
        for (int a = lane; a < kA; a += 32) {
            float v = bf2f(row[a]) + sdec[a];
            v = v > 0.0f ? v : 0.0f;
            s += v * sw[a];
        }
        for (int off = 16; off; off >>= 1) s += __shfl_down(s, off, 32);
        if (lane == 0) se[p] = s + attb[0];
    }
    __syncthreads();

    if (wave == 0) {  // softmax over 196 values
        float mx = -3.4e38f;
        for (int p = lane; p < kP; p += 32) mx = fmaxf(mx, se[p]);
        for (int off = 16; off; off >>= 1) mx = fmaxf(mx, __shfl_down(mx, off, 32));
        mx = __shfl(mx, 0, 32);
        float sum = 0.0f;
        for (int p = lane; p < kP; p += 32) { float ev = __expf(se[p] - mx); se[p] = ev; sum += ev; }
        for (int off = 16; off; off >>= 1) sum += __shfl_down(sum, off, 32);
        sum = __shfl(sum, 0, 32);
        const float inv = 1.0f / sum;
        for (int p = lane; p < kP; p += 32) se[p] *= inv;
    }
    __syncthreads();

    const float mask = (dlen[b] - 1 > t) ? 1.0f : 0.0f;
    for (int p = tid; p < kP; p += 256) {
        const float av = se[p];
        alpha[b * kP + p] = av;
        out[OFF_ALPHA + ((size_t)b * kT + t) * kP + p] = av * mask;
    }
}

// awe[b][e] = sum_p alpha[b][p] * enc[b][p][e]; grid (B, E/256)
__global__ void k_awe(const unsigned short* __restrict__ encbf,
                      const float* __restrict__ alpha, float* __restrict__ awe) {
    __shared__ float sa[kP];
    const int b = blockIdx.x;
    for (int i = threadIdx.x; i < kP; i += 256) sa[i] = alpha[b * kP + i];
    __syncthreads();
    const int e = blockIdx.y * 256 + threadIdx.x;
    const unsigned short* base = encbf + (size_t)b * kP * kE + e;
    float s = 0.0f;
    for (int p = 0; p < kP; ++p) s += sa[p] * bf2f(base[(size_t)p * kE]);
    awe[b * kE + e] = s;
}

// build xh = [emb[word] | sigmoid(gate)*awe | h] in bf16; grid (B, XH/256)
__global__ void k_combine(const float* __restrict__ gatepre, const float* __restrict__ awe,
                          const unsigned short* __restrict__ embbf,
                          const int* __restrict__ caps, const int* __restrict__ prev, int t,
                          const float* __restrict__ h, unsigned short* __restrict__ xh) {
    const int b = blockIdx.x;
    const int col = blockIdx.y * 256 + threadIdx.x;
    unsigned short o;
    if (col < kM) {
        const int w = (t == 0) ? caps[b * kCap] : prev[b];
        o = embbf[(size_t)w * kM + col];
    } else if (col < kM + kE) {
        const int e = col - kM;
        const float g = sigmoidf(gatepre[b * kE + e]);
        o = f2bf(g * awe[b * kE + e]);
    } else {
        o = f2bf(h[b * kD + (col - kM - kE)]);
    }
    xh[(size_t)b * kXH + col] = o;
}

// LSTM cell pointwise; grid (B, D/256)
__global__ void k_lstm(const float* __restrict__ gates, const int* __restrict__ dlen, int t,
                       float* __restrict__ h, float* __restrict__ c,
                       unsigned short* __restrict__ hbf, unsigned short* __restrict__ hnewbf) {
    const int b = blockIdx.x;
    const int d = blockIdx.y * 256 + threadIdx.x;
    const float* g = gates + (size_t)b * kG;
    const float ig = sigmoidf(g[d]);
    const float fg = sigmoidf(g[kD + d]);
    const float gg = tanhf(g[2 * kD + d]);
    const float og = sigmoidf(g[3 * kD + d]);
    const float cn = fg * c[b * kD + d] + ig * gg;
    const float hn = og * tanhf(cn);
    hnewbf[b * kD + d] = f2bf(hn);                   // raw h_new for vocab GEMM
    if (dlen[b] - 1 > t) {                           // masked carry update
        h[b * kD + d] = hn;
        c[b * kD + d] = cn;
        hbf[b * kD + d] = f2bf(hn);
    }
}

// masked store of predictions + argmax over raw preds; one block per batch row
__global__ __launch_bounds__(256) void k_finalize(
    const float* __restrict__ preds, const int* __restrict__ dlen, int t,
    float* __restrict__ out, int* __restrict__ prev) {
    __shared__ float sv[256];
    __shared__ int   si[256];
    const int b = blockIdx.x, tid = threadIdx.x;
    const float* pr = preds + (size_t)b * kV;
    const float mask = (dlen[b] - 1 > t) ? 1.0f : 0.0f;
    float* orow = out + ((size_t)b * kT + t) * kV;
    float best = -3.4e38f;
    int bi = 0;
    for (int v = tid; v < kV; v += 256) {
        const float val = pr[v];
        orow[v] = val * mask;
        if (val > best) { best = val; bi = v; }
    }
    sv[tid] = best; si[tid] = bi;
    __syncthreads();
    for (int s = 128; s; s >>= 1) {
        if (tid < s) {
            if (sv[tid + s] > sv[tid] ||
                (sv[tid + s] == sv[tid] && si[tid + s] < si[tid])) {
                sv[tid] = sv[tid + s]; si[tid] = si[tid + s];
            }
        }
        __syncthreads();
    }
    if (tid == 0) prev[b] = si[0];
}

// ---------------------------------------------------------------------------
// Host driver
// ---------------------------------------------------------------------------
extern "C" void kernel_launch(void* const* d_in, const int* in_sizes, int n_in,
                              void* d_out, int out_size, void* d_ws, size_t ws_size,
                              hipStream_t stream) {
    (void)in_sizes; (void)n_in; (void)out_size; (void)ws_size;

    const float* encoder_out = (const float*)d_in[0];
    const int*   captions    = (const int*)d_in[1];
    const int*   dlen        = (const int*)d_in[2];
    const int*   inds        = (const int*)d_in[3];
    const float* enc_att_w   = (const float*)d_in[4];
    const float* enc_att_b   = (const float*)d_in[5];
    const float* dec_att_w   = (const float*)d_in[6];
    const float* dec_att_b   = (const float*)d_in[7];
    const float* att_w       = (const float*)d_in[8];
    const float* att_b       = (const float*)d_in[9];
    const float* init_h_w    = (const float*)d_in[10];
    const float* init_h_b    = (const float*)d_in[11];
    const float* init_c_w    = (const float*)d_in[12];
    const float* init_c_b    = (const float*)d_in[13];
    const float* f_beta_w    = (const float*)d_in[14];
    const float* f_beta_b    = (const float*)d_in[15];
    const float* w_ih        = (const float*)d_in[16];
    const float* b_ih        = (const float*)d_in[17];
    const float* w_hh        = (const float*)d_in[18];
    const float* b_hh        = (const float*)d_in[19];
    const float* emb         = (const float*)d_in[20];
    const float* fc_w        = (const float*)d_in[21];
    const float* fc_b        = (const float*)d_in[22];
    float* out = (float*)d_out;

    // ---- workspace carve-up (256B aligned) ----
    char* w = (char*)d_ws;
    size_t off = 0;
    auto take = [&](size_t bytes) -> void* {
        void* p = w + off;
        off = (off + bytes + 255) & ~(size_t)255;
        return p;
    };
    unsigned short* enc_bf  = (unsigned short*)take((size_t)kB * kP * kE * 2);   // 51.4 MB
    unsigned short* encattT = (unsigned short*)take((size_t)kA * kE * 2);
    unsigned short* encatt  = (unsigned short*)take((size_t)kB * kP * kA * 2);   // 12.8 MB
    unsigned short* decattT = (unsigned short*)take((size_t)kA * kD * 2);
    unsigned short* fbetaT  = (unsigned short*)take((size_t)kE * kD * 2);
    unsigned short* wcat    = (unsigned short*)take((size_t)kG * kXH * 2);       // 12.6 MB
    unsigned short* fcT     = (unsigned short*)take((size_t)kV * kD * 2);        // 10.2 MB
    unsigned short* embbf   = (unsigned short*)take((size_t)kV * kM * 2);        // 10.2 MB
    float* bg       = (float*)take((size_t)kG * 4);
    float* meanbuf  = (float*)take((size_t)kB * kE * 4);
    float* hbuf     = (float*)take((size_t)kB * kD * 4);
    float* cbuf     = (float*)take((size_t)kB * kD * 4);
    unsigned short* hbf    = (unsigned short*)take((size_t)kB * kD * 2);
    unsigned short* hnewbf = (unsigned short*)take((size_t)kB * kD * 2);
    float* decbuf   = (float*)take((size_t)kB * kA * 4);
    float* alpha    = (float*)take((size_t)kB * kP * 4);
    float* awebuf   = (float*)take((size_t)kB * kE * 4);
    float* gatepre  = (float*)take((size_t)kB * kE * 4);
    unsigned short* xh = (unsigned short*)take((size_t)kB * kXH * 2);
    float* gatespart= (float*)take((size_t)kKS * kB * kG * 4);                   // split-K partials
    float* gatesbuf = (float*)take((size_t)kB * kG * 4);
    float* predsbuf = (float*)take((size_t)kB * kV * 4);                         // 2.56 MB
    int*   prev     = (int*)take((size_t)kB * 4);

    auto cdiv = [](int a, int b) { return (a + b - 1) / b; };

    // ---- one-time weight prep (re-run every call; deterministic) ----
    k_transpose_f2bf<<<cdiv(kE * kA, 256), 256, 0, stream>>>(enc_att_w, encattT, kE, kA);
    k_transpose_f2bf<<<cdiv(kD * kA, 256), 256, 0, stream>>>(dec_att_w, decattT, kD, kA);
    k_transpose_f2bf<<<cdiv(kD * kE, 256), 256, 0, stream>>>(f_beta_w, fbetaT, kD, kE);
    k_transpose_f2bf<<<cdiv(kD * kV, 256), 256, 0, stream>>>(fc_w, fcT, kD, kV);
    k_f2bf<<<cdiv(kV * kM, 256), 256, 0, stream>>>(emb, embbf, kV * kM);
    k_f2bf<<<cdiv(kB * kP * kE, 256), 256, 0, stream>>>(encoder_out, enc_bf, kB * kP * kE);
    k_f2bf_sub<<<cdiv(kG * (kM + kE), 256), 256, 0, stream>>>(w_ih, wcat, kG, kM + kE, kXH, 0);
    k_f2bf_sub<<<cdiv(kG * kD, 256), 256, 0, stream>>>(w_hh, wcat, kG, kD, kXH, kM + kE);
    k_vadd<<<cdiv(kG, 256), 256, 0, stream>>>(b_ih, b_hh, bg, kG);

    // h0 / c0
    k_mean<<<dim3(kB, kE / 256), 256, 0, stream>>>(encoder_out, meanbuf);
    k_init<<<dim3(kB, kD / 256), 256, 0, stream>>>(meanbuf, init_h_w, init_h_b,
                                                   init_c_w, init_c_b, hbuf, cbuf, hbf);

    // enc_att = encoder_out @ enc_att_w + b  (12544 x 512, K=2048) -> bf16
    k_gemm_bf16<true><<<dim3(kB * kP / 16, 1), 256, 0, stream>>>(
        enc_bf, kE, encattT, kE, enc_att_b, encatt, kA, kA, kE);

    k_dlinds<<<1, 64, 0, stream>>>(dlen, inds, out);

    // ---- recurrent steps ----
    for (int t = 0; t < kT; ++t) {
        // dec = h @ dec_att_w + b           (64 x 512, K=512)
        k_gemm_skinny<false><<<dim3(kA / 128, 1), 256, 0, stream>>>(
            hbf, kD, decattT, kD, dec_att_b, decbuf, kA, kA, kD, kD);
        // e = relu(enc_att + dec) @ att_w ; alpha = softmax
        k_att<<<kB, 256, 0, stream>>>(encatt, decbuf, att_w, att_b, dlen, t, alpha, out);
        // awe = sum_p alpha * enc
        k_awe<<<dim3(kB, kE / 256), 256, 0, stream>>>(enc_bf, alpha, awebuf);
        // gate = h @ f_beta_w + b           (64 x 2048, K=512)
        k_gemm_skinny<false><<<dim3(kE / 128, 1), 256, 0, stream>>>(
            hbf, kD, fbetaT, kD, f_beta_b, gatepre, kE, kE, kD, kD);
        // xh = [emb[word] | sigmoid(gate)*awe | h]
        k_combine<<<dim3(kB, kXH / 256), 256, 0, stream>>>(
            gatepre, awebuf, embbf, captions, prev, t, hbuf, xh);
        // gates = xh @ [w_ih|w_hh]^T : split-K=4 partials, then reduce(+bias)
        k_gemm_skinny<true><<<dim3(kG / 128, kKS), 256, 0, stream>>>(
            xh, kXH, wcat, kXH, nullptr, gatespart, kG, kG, kXH, kXH / kKS);
        k_reduce<<<cdiv(kB * kG, 256), 256, 0, stream>>>(
            gatespart, bg, gatesbuf, kB * kG, kG, kKS);
        // LSTM cell + masked carry update
        k_lstm<<<dim3(kB, kD / 256), 256, 0, stream>>>(
            gatesbuf, dlen, t, hbuf, cbuf, hbf, hnewbf);
        // preds = h_new @ fc_w + b          (64 x 10000, K=512)
        k_gemm_skinny<false><<<dim3(cdiv(kV, 128), 1), 256, 0, stream>>>(
            hnewbf, kD, fcT, kD, fc_b, predsbuf, kV, kV, kD, kD);
        // masked store + argmax -> prev
        k_finalize<<<kB, 256, 0, stream>>>(predsbuf, dlen, t, out, prev);
    }
}
